// GraspPointPredictor_3702261809766
// MI455X (gfx1250) — compile-verified, matrix operand order checked by reading, not run on hardware
//
#include <hip/hip_runtime.h>
#include <hip/hip_bf16.h>

// ---------------------------------------------------------------------------
// Types for CDNA5 WMMA (wave32): v_wmma_f32_16x16x32_f16
// ---------------------------------------------------------------------------
typedef _Float16 v16h __attribute__((ext_vector_type(16)));
typedef float    v8f  __attribute__((ext_vector_type(8)));

// Per-lane fragment loader for the ISA 16-bit A-matrix 16x32 layout:
//   elements 0..7  = K s..s+7, elements 8..15 = K s+16..s+23, s = (lane>=16)*8.
// Works for global or LDS pointers (addrspace inferred after inlining).
// B fragments use the identical pattern applied to W^T rows (lane = n).
__device__ __forceinline__ v16h load_frag16(const _Float16* __restrict__ p) {
    v16h r;
    *(uint4*)(&r)       = *(const uint4*)(p);
    *((uint4*)(&r) + 1) = *(const uint4*)(p + 16);
    return r;
}

#define WMMA_F16(a, b, c) \
    __builtin_amdgcn_wmma_f32_16x16x32_f16(false, (a), false, (b), (short)0, (c), false, false)

// ---------------------------------------------------------------------------
// Generic WMMA GEMM (used for fp3 / fp2 stages):
//   out[row][n] = act( A[row][:] . WT[n][:] + addend )
//   addend = rowconst ? rowconst[(row/rows_per_batch)*N + n] : bias[n]
// Wave -> 16x32 tile; block = 8 waves = 128 rows x 32 cols.
// Grid: (N/32, rows/128), block 256.
// ---------------------------------------------------------------------------
__global__ __launch_bounds__(256)
void wmma_gemm_f16(const _Float16* __restrict__ A, const _Float16* __restrict__ WT,
                   const float* __restrict__ bias, const float* __restrict__ rowconst,
                   int K, int N, int rows_per_batch, int relu,
                   _Float16* __restrict__ out)
{
    const int wave    = threadIdx.x >> 5;
    const int lane    = threadIdx.x & 31;
    const int rowBase = blockIdx.y * 128 + wave * 16;
    const int colBase = blockIdx.x * 32;

    const int m = lane & 15;
    const int s = (lane >> 4) << 3;          // 0 or 8

    const _Float16* pA  = A  + (size_t)(rowBase + m)       * K + s;
    const _Float16* pB0 = WT + (size_t)(colBase + m)       * K + s;
    const _Float16* pB1 = WT + (size_t)(colBase + 16 + m)  * K + s;

    v8f c0 = {}; v8f c1 = {};
    for (int k0 = 0; k0 < K; k0 += 32) {
        v16h a  = load_frag16(pA  + k0);
        v16h b0 = load_frag16(pB0 + k0);
        v16h b1 = load_frag16(pB1 + k0);
        c0 = WMMA_F16(a, b0, c0);
        c1 = WMMA_F16(a, b1, c1);
    }

    const int n0   = colBase + (lane & 15);
    const int mOff = (lane >> 4) << 3;
    float add0, add1;
    if (rowconst) {
        const int b = rowBase / rows_per_batch;   // 16-row tiles never straddle a batch
        add0 = rowconst[b * N + n0];
        add1 = rowconst[b * N + n0 + 16];
    } else {
        add0 = bias[n0];
        add1 = bias[n0 + 16];
    }
#pragma unroll
    for (int r = 0; r < 8; ++r) {
        const int row = rowBase + mOff + r;
        float v0 = c0[r] + add0;
        float v1 = c1[r] + add1;
        if (relu) { v0 = fmaxf(v0, 0.f); v1 = fmaxf(v1, 0.f); }
        out[(size_t)row * N + n0]      = (_Float16)v0;
        out[(size_t)row * N + n0 + 16] = (_Float16)v1;
    }
}

// ---------------------------------------------------------------------------
// Fused 6-layer tail over the 32768-row activation stream:
//   L0: A4h[32768,160] x Wt0 -> relu         (fp1_w1, K=160)
//   L1: x Wt1 -> relu                        (fp1_w2)
//   L2: x Wt2                                (fp1_w3)
//   L3: x Wt3 + ch[b] -> relu                (head_w1, af folded into ch)
//   L4: x Wt4 -> relu                        (head_w2)
//   L5: dot(., w3) + b3 -> out (f32)         (head_w3)
// Block = 256 threads = 8 waves = 64 rows; wave = 16 rows x 64 cols.
// Inter-layer activations ping-pong in LDS (stride 136 halves, conflict-free
// two-phase b128 access). Weights read from global (tiny, L2-resident).
// ---------------------------------------------------------------------------
#define TSTR 136

__device__ __forceinline__ void layer128_lds(
    const _Float16* actIn, _Float16* actOut,
    const _Float16* __restrict__ Wt, const float* __restrict__ addv, int relu,
    int rowL, int colB, int m, int s, int n0, int mOff)
{
    v8f c[4] = {};
    const _Float16* pA = actIn + (rowL + m) * TSTR + s;
#pragma unroll
    for (int k0 = 0; k0 < 128; k0 += 32) {
        v16h a = load_frag16(pA + k0);
#pragma unroll
        for (int t = 0; t < 4; ++t) {
            v16h b = load_frag16(Wt + (size_t)(colB + t * 16 + m) * 128 + k0 + s);
            c[t] = WMMA_F16(a, b, c[t]);
        }
    }
#pragma unroll
    for (int t = 0; t < 4; ++t) {
        const int col = colB + t * 16 + n0;
        const float add = addv[col];
#pragma unroll
        for (int r = 0; r < 8; ++r) {
            float v = c[t][r] + add;
            if (relu) v = fmaxf(v, 0.f);
            actOut[(rowL + mOff + r) * TSTR + col] = (_Float16)v;
        }
    }
}

__global__ __launch_bounds__(256)
void fused_tail_kernel(const _Float16* __restrict__ A4h,
                       const _Float16* __restrict__ Wt0, const float* __restrict__ b0,
                       const _Float16* __restrict__ Wt1, const float* __restrict__ b1,
                       const _Float16* __restrict__ Wt2, const float* __restrict__ b2,
                       const _Float16* __restrict__ Wt3, const float* __restrict__ chv,
                       const _Float16* __restrict__ Wt4, const float* __restrict__ b4,
                       const float* __restrict__ w3, const float* __restrict__ b3f,
                       float* __restrict__ out)
{
    __shared__ _Float16 act0[64 * TSTR];
    __shared__ _Float16 act1[64 * TSTR];

    const int wave = threadIdx.x >> 5;
    const int lane = threadIdx.x & 31;
    const int rowL = (wave >> 1) * 16;        // 0,16,32,48
    const int colB = (wave & 1) * 64;         // 0 or 64
    const int rowG = blockIdx.x * 64;

    const int m    = lane & 15;
    const int s    = (lane >> 4) << 3;
    const int n0   = lane & 15;
    const int mOff = (lane >> 4) << 3;

    // ---- L0: global A4h (K=160) -> act0 ----
    {
        v8f c[4] = {};
        const _Float16* pA = A4h + (size_t)(rowG + rowL + m) * 160 + s;
#pragma unroll
        for (int k0 = 0; k0 < 160; k0 += 32) {
            v16h a = load_frag16(pA + k0);
#pragma unroll
            for (int t = 0; t < 4; ++t) {
                v16h b = load_frag16(Wt0 + (size_t)(colB + t * 16 + m) * 160 + k0 + s);
                c[t] = WMMA_F16(a, b, c[t]);
            }
        }
#pragma unroll
        for (int t = 0; t < 4; ++t) {
            const int col = colB + t * 16 + n0;
            const float add = b0[col];
#pragma unroll
            for (int r = 0; r < 8; ++r)
                act0[(rowL + mOff + r) * TSTR + col] = (_Float16)fmaxf(c[t][r] + add, 0.f);
        }
    }
    __syncthreads();

    // ---- L1..L4: LDS ping-pong ----
    layer128_lds(act0, act1, Wt1, b1, 1, rowL, colB, m, s, n0, mOff);
    __syncthreads();
    layer128_lds(act1, act0, Wt2, b2, 0, rowL, colB, m, s, n0, mOff);
    __syncthreads();
    const float* addv3 = chv + (rowG >> 12) * 128;   // per-batch row-constant (incl. head_b1)
    layer128_lds(act0, act1, Wt3, addv3, 1, rowL, colB, m, s, n0, mOff);
    __syncthreads();
    layer128_lds(act1, act0, Wt4, b4, 1, rowL, colB, m, s, n0, mOff);
    __syncthreads();

    // ---- L5: 128 -> 1 dot per row ----
    if (threadIdx.x < 64) {
        const _Float16* rw = act0 + threadIdx.x * TSTR;
        float acc = b3f[0];
#pragma unroll 8
        for (int k = 0; k < 128; ++k) acc += (float)rw[k] * w3[k];
        out[rowG + threadIdx.x] = acc;
    }
}

// ---------------------------------------------------------------------------
// Weight prep: dst[n][k] (f16, [N][Kpad]) = src[k][n] (f32, [K][N]); zero pad.
// ---------------------------------------------------------------------------
__global__ void conv_w_kernel(const float* __restrict__ src, _Float16* __restrict__ dst,
                              int K, int N, int Kpad)
{
    int t = blockIdx.x * blockDim.x + threadIdx.x;
    if (t >= N * Kpad) return;
    int n = t / Kpad, k = t % Kpad;
    dst[t] = (k < K) ? (_Float16)src[(size_t)k * N + n] : (_Float16)0.f;
}

__global__ void conv_act_kernel(const float* __restrict__ src, _Float16* __restrict__ dst, int n)
{
    int t = blockIdx.x * blockDim.x + threadIdx.x;
    if (t < n) dst[t] = (_Float16)src[t];
}

// ---------------------------------------------------------------------------
// c3[b][n] = sum_k x3[b][k] * fp3_w1[k][n] + fp3_b1[n]     (k < 1024, n < 256)
// (fp3 k=1 interp from a single source point is an exact broadcast of x3)
// ---------------------------------------------------------------------------
__global__ void c3_kernel(const float* __restrict__ x3, const float* __restrict__ w1,
                          const float* __restrict__ b1, float* __restrict__ c3)
{
    int t = blockIdx.x * blockDim.x + threadIdx.x;      // 8*256
    if (t >= 8 * 256) return;
    int b = t >> 8, n = t & 255;
    const float* xr = x3 + (size_t)b * 1024;
    float acc = b1[n];
    for (int k = 0; k < 1024; ++k) acc += xr[k] * w1[(size_t)k * 256 + n];
    c3[t] = acc;
}

// ---------------------------------------------------------------------------
// Approach-point encoder folded into head layer-1 per-batch constant:
// t = relu(ap @ enc_w1 + enc_b1); af = t @ enc_w2 + enc_b2
// ch[b][n] = head_b1[n] + sum_j af[j] * head_w1[128+j][n]
// ---------------------------------------------------------------------------
__global__ void ch_kernel(const float* __restrict__ ap,
                          const float* __restrict__ ew1, const float* __restrict__ eb1,
                          const float* __restrict__ ew2, const float* __restrict__ eb2,
                          const float* __restrict__ hw1, const float* __restrict__ hb1,
                          float* __restrict__ ch)
{
    int t = blockIdx.x * blockDim.x + threadIdx.x;      // 8*128
    if (t >= 8 * 128) return;
    int b = t >> 7, n = t & 127;
    float a0 = ap[b * 3 + 0], a1 = ap[b * 3 + 1], a2 = ap[b * 3 + 2];
    float tb[32];
#pragma unroll
    for (int j = 0; j < 32; ++j)
        tb[j] = fmaxf(a0 * ew1[j] + a1 * ew1[32 + j] + a2 * ew1[64 + j] + eb1[j], 0.f);
    float acc = hb1[n];
    for (int j = 0; j < 32; ++j) {
        float af = eb2[j];
#pragma unroll
        for (int i = 0; i < 32; ++i) af += tb[i] * ew2[i * 32 + j];
        acc += af * hw1[(size_t)(128 + j) * 128 + n];
    }
    ch[t] = acc;
}

// ---------------------------------------------------------------------------
// KNN (k=3) inverse-distance-squared interpolation kernels.
// Strict '<' keeps the earliest index on ties (matches jax top_k stability).
// ---------------------------------------------------------------------------
// sa2 (256 src, 256 ch) -> sa1 (1024 queries); out row = [interp(256) | x1(128)]
__global__ void knn2_kernel(const float* __restrict__ pos2, const float* __restrict__ pos1,
                            const float* __restrict__ x1, const _Float16* __restrict__ feat,
                            _Float16* __restrict__ out)
{
    int gid = blockIdx.x * blockDim.x + threadIdx.x;    // 8*1024
    if (gid >= 8 * 1024) return;
    int b = gid >> 10;
    float qx = pos1[gid * 3 + 0], qy = pos1[gid * 3 + 1], qz = pos1[gid * 3 + 2];
    const float* src = pos2 + (size_t)b * 256 * 3;
    float d0 = 3.4e38f, d1 = 3.4e38f, d2 = 3.4e38f;
    int   i0 = 0, i1 = 0, i2 = 0;
    for (int i = 0; i < 256; ++i) {
        float dx = src[i * 3 + 0] - qx, dy = src[i * 3 + 1] - qy, dz = src[i * 3 + 2] - qz;
        float d = dx * dx + dy * dy + dz * dz;
        if (d < d0)      { d2 = d1; i2 = i1; d1 = d0; i1 = i0; d0 = d; i0 = i; }
        else if (d < d1) { d2 = d1; i2 = i1; d1 = d;  i1 = i; }
        else if (d < d2) { d2 = d;  i2 = i; }
    }
    float w0 = 1.f / fmaxf(d0, 1e-16f);
    float w1 = 1.f / fmaxf(d1, 1e-16f);
    float w2 = 1.f / fmaxf(d2, 1e-16f);
    float inv = 1.f / (w0 + w1 + w2);
    w0 *= inv; w1 *= inv; w2 *= inv;
    const _Float16* f0 = feat + (size_t)(b * 256 + i0) * 256;
    const _Float16* f1 = feat + (size_t)(b * 256 + i1) * 256;
    const _Float16* f2 = feat + (size_t)(b * 256 + i2) * 256;
    _Float16* o = out + (size_t)gid * 384;
    for (int c = 0; c < 256; ++c)
        o[c] = (_Float16)(w0 * (float)f0[c] + w1 * (float)f1[c] + w2 * (float)f2[c]);
    const float* xr = x1 + (size_t)gid * 128;
    for (int c = 0; c < 128; ++c) o[256 + c] = (_Float16)xr[c];
}

// sa1 (1024 src, 128 ch) -> sa0 (4096 queries); out row = [interp(128)|pos0(3)|pad(29)]
__global__ void knn1_kernel(const float* __restrict__ pos1, const float* __restrict__ pos0,
                            const _Float16* __restrict__ feat, _Float16* __restrict__ out)
{
    int gid = blockIdx.x * blockDim.x + threadIdx.x;    // 8*4096
    if (gid >= 8 * 4096) return;
    int b = gid >> 12;
    float qx = pos0[gid * 3 + 0], qy = pos0[gid * 3 + 1], qz = pos0[gid * 3 + 2];
    const float* src = pos1 + (size_t)b * 1024 * 3;
    float d0 = 3.4e38f, d1 = 3.4e38f, d2 = 3.4e38f;
    int   i0 = 0, i1 = 0, i2 = 0;
    for (int i = 0; i < 1024; ++i) {
        float dx = src[i * 3 + 0] - qx, dy = src[i * 3 + 1] - qy, dz = src[i * 3 + 2] - qz;
        float d = dx * dx + dy * dy + dz * dz;
        if (d < d0)      { d2 = d1; i2 = i1; d1 = d0; i1 = i0; d0 = d; i0 = i; }
        else if (d < d1) { d2 = d1; i2 = i1; d1 = d;  i1 = i; }
        else if (d < d2) { d2 = d;  i2 = i; }
    }
    float w0 = 1.f / fmaxf(d0, 1e-16f);
    float w1 = 1.f / fmaxf(d1, 1e-16f);
    float w2 = 1.f / fmaxf(d2, 1e-16f);
    float inv = 1.f / (w0 + w1 + w2);
    w0 *= inv; w1 *= inv; w2 *= inv;
    const _Float16* f0 = feat + (size_t)(b * 1024 + i0) * 128;
    const _Float16* f1 = feat + (size_t)(b * 1024 + i1) * 128;
    const _Float16* f2 = feat + (size_t)(b * 1024 + i2) * 128;
    _Float16* o = out + (size_t)gid * 160;
    for (int c = 0; c < 128; ++c)
        o[c] = (_Float16)(w0 * (float)f0[c] + w1 * (float)f1[c] + w2 * (float)f2[c]);
    o[128] = (_Float16)qx; o[129] = (_Float16)qy; o[130] = (_Float16)qz;
#pragma unroll
    for (int c = 131; c < 160; ++c) o[c] = (_Float16)0.f;
}

// ---------------------------------------------------------------------------
// Host launcher
// ---------------------------------------------------------------------------
extern "C" void kernel_launch(void* const* d_in, const int* in_sizes, int n_in,
                              void* d_out, int out_size, void* d_ws, size_t ws_size,
                              hipStream_t stream)
{
    const float* pos0   = (const float*)d_in[0];
    const float* x1     = (const float*)d_in[1];
    const float* pos1   = (const float*)d_in[2];
    const float* x2     = (const float*)d_in[3];
    const float* pos2   = (const float*)d_in[4];
    const float* x3     = (const float*)d_in[5];
    /* pos3 = d_in[6] unused: k=1 interp from one source is a broadcast */
    const float* ap     = (const float*)d_in[7];
    const float* fp3_w1 = (const float*)d_in[8];
    const float* fp3_b1 = (const float*)d_in[9];
    const float* fp3_w2 = (const float*)d_in[10];
    const float* fp3_b2 = (const float*)d_in[11];
    const float* fp2_w1 = (const float*)d_in[12];
    const float* fp2_b1 = (const float*)d_in[13];
    const float* fp2_w2 = (const float*)d_in[14];
    const float* fp2_b2 = (const float*)d_in[15];
    const float* fp1_w1 = (const float*)d_in[16];
    const float* fp1_b1 = (const float*)d_in[17];
    const float* fp1_w2 = (const float*)d_in[18];
    const float* fp1_b2 = (const float*)d_in[19];
    const float* fp1_w3 = (const float*)d_in[20];
    const float* fp1_b3 = (const float*)d_in[21];
    const float* enc_w1 = (const float*)d_in[22];
    const float* enc_b1 = (const float*)d_in[23];
    const float* enc_w2 = (const float*)d_in[24];
    const float* enc_b2 = (const float*)d_in[25];
    const float* head_w1 = (const float*)d_in[26];
    const float* head_b1 = (const float*)d_in[27];
    const float* head_w2 = (const float*)d_in[28];
    const float* head_b2 = (const float*)d_in[29];
    const float* head_w3 = (const float*)d_in[30];
    const float* head_b3 = (const float*)d_in[31];
    float* out = (float*)d_out;

    // ---- workspace carve-up (256B aligned) ----
    char* ws = (char*)d_ws;
    size_t off = 0;
    auto alloc = [&](size_t bytes) { size_t o = off; off += (bytes + 255) & ~(size_t)255; return o; };
    _Float16* w_fp3w1t = (_Float16*)(ws + alloc(256 * 256 * 2));
    _Float16* w_fp3w2t = (_Float16*)(ws + alloc(256 * 256 * 2));
    _Float16* w_fp2w1t = (_Float16*)(ws + alloc(256 * 384 * 2));
    _Float16* w_fp2w2t = (_Float16*)(ws + alloc(128 * 256 * 2));
    _Float16* w_fp1w1t = (_Float16*)(ws + alloc(128 * 160 * 2));
    _Float16* w_fp1w2t = (_Float16*)(ws + alloc(128 * 128 * 2));
    _Float16* w_fp1w3t = (_Float16*)(ws + alloc(128 * 128 * 2));
    _Float16* w_hw1t   = (_Float16*)(ws + alloc(128 * 128 * 2));
    _Float16* w_hw2t   = (_Float16*)(ws + alloc(128 * 128 * 2));
    float*    c3       = (float*)   (ws + alloc(8 * 256 * 4));
    float*    chv      = (float*)   (ws + alloc(8 * 128 * 4));
    _Float16* x2h      = (_Float16*)(ws + alloc((size_t)2048 * 256 * 2));
    _Float16* h1h      = (_Float16*)(ws + alloc((size_t)2048 * 256 * 2));
    _Float16* h2h      = (_Float16*)(ws + alloc((size_t)2048 * 256 * 2));
    _Float16* A2h      = (_Float16*)(ws + alloc((size_t)8192 * 384 * 2));
    _Float16* h3h      = (_Float16*)(ws + alloc((size_t)8192 * 256 * 2));
    _Float16* h4h      = (_Float16*)(ws + alloc((size_t)8192 * 128 * 2));
    _Float16* A4h      = (_Float16*)(ws + alloc((size_t)32768 * 160 * 2));
    (void)ws_size; (void)in_sizes; (void)n_in; (void)out_size;

    auto cw = [&](const float* src, _Float16* dst, int K, int N, int Kpad) {
        int n = N * Kpad;
        conv_w_kernel<<<(n + 255) / 256, 256, 0, stream>>>(src, dst, K, N, Kpad);
    };

    // ---- weight prep (f16, transposed, K-padded) ----
    cw(fp3_w1 + (size_t)1024 * 256, w_fp3w1t, 256, 256, 256);   // x2 part of fp3_w1
    cw(fp3_w2, w_fp3w2t, 256, 256, 256);
    cw(fp2_w1, w_fp2w1t, 384, 256, 384);
    cw(fp2_w2, w_fp2w2t, 256, 128, 256);
    cw(fp1_w1, w_fp1w1t, 131, 128, 160);                        // pad 131 -> 160
    cw(fp1_w2, w_fp1w2t, 128, 128, 128);
    cw(fp1_w3, w_fp1w3t, 128, 128, 128);
    cw(head_w1, w_hw1t, 128, 128, 128);                         // h part only
    cw(head_w2, w_hw2t, 128, 128, 128);

    // ---- per-batch constants ----
    c3_kernel<<<8, 256, 0, stream>>>(x3, fp3_w1, fp3_b1, c3);
    ch_kernel<<<4, 256, 0, stream>>>(ap, enc_w1, enc_b1, enc_w2, enc_b2, head_w1, head_b1, chv);

    // ---- activations f32 -> f16 ----
    conv_act_kernel<<<2048, 256, 0, stream>>>(x2, x2h, 2048 * 256);

    // ---- fp3 ----
    wmma_gemm_f16<<<dim3(256 / 32, 2048 / 128), 256, 0, stream>>>(
        x2h, w_fp3w1t, nullptr, c3, 256, 256, 256, 1, h1h);
    wmma_gemm_f16<<<dim3(256 / 32, 2048 / 128), 256, 0, stream>>>(
        h1h, w_fp3w2t, fp3_b2, nullptr, 256, 256, 1, 0, h2h);

    // ---- fp2 ----
    knn2_kernel<<<32, 256, 0, stream>>>(pos2, pos1, x1, h2h, A2h);
    wmma_gemm_f16<<<dim3(256 / 32, 8192 / 128), 256, 0, stream>>>(
        A2h, w_fp2w1t, fp2_b1, nullptr, 384, 256, 1, 1, h3h);
    wmma_gemm_f16<<<dim3(128 / 32, 8192 / 128), 256, 0, stream>>>(
        h3h, w_fp2w2t, fp2_b2, nullptr, 256, 128, 1, 0, h4h);

    // ---- fp1 interp ----
    knn1_kernel<<<128, 256, 0, stream>>>(pos1, pos0, h4h, A4h);

    // ---- fused fp1 MLP + head (6 layers, LDS-resident activations) ----
    fused_tail_kernel<<<32768 / 64, 256, 0, stream>>>(
        A4h,
        w_fp1w1t, fp1_b1,
        w_fp1w2t, fp1_b2,
        w_fp1w3t, fp1_b3,
        w_hw1t,   chv,
        w_hw2t,   head_b2,
        head_w3,  head_b3,
        out);
}
